// T5Backbone_22119081574629
// MI455X (gfx1250) — matrix operand order repaired and checked by reading, not compiled, not run on hardware
//
#include <hip/hip_runtime.h>

// T5 decoder backbone: L=6, H=1024, NH=16, D=64, F=4096, B=4, S=512 (fp32 in/out).
// GEMMs: v_wmma_f32_16x16x32_f16, double-buffered LDS tiles, b128 staging.
// Attention: flash-style online softmax, WMMA for QK^T and PV.

typedef __attribute__((ext_vector_type(16))) _Float16 v16h;
typedef __attribute__((ext_vector_type(8)))  float    v8f;

#define LBL 6
#define HDIM 1024
#define FDIM 4096
#define NHEAD 16
#define DHEAD 64
#define BSZ 4
#define SLEN 512
#define MTOK (BSZ * SLEN)   // 2048

union FragH { v16h h; unsigned u[8]; };

// A-matrix fragment (16x32 f16, M x K): lane r holds row r (both halves).
// K interleave per ISA table: half0 -> {0..7,16..23}, half1 -> {8..15,24..31}.
__device__ __forceinline__ v16h load_frag_a(const _Float16* rowp, int kofs, int hf) {
  FragH f;
#pragma unroll
  for (int g = 0; g < 8; ++g) {
    int k = kofs + ((g & 3) << 1) + ((g & 4) << 2) + (hf << 3);
    f.u[g] = *(const unsigned*)(rowp + k);
  }
  return f.h;
}

// B-matrix fragment (32x16 f16, K x N): lane r holds column r; half0 K={0..7,16..23},
// half1 K={8..15,24..31}. colp points at column r in a [N][K]-transposed LDS tile.
__device__ __forceinline__ v16h load_frag_b(const _Float16* colp, int kofs, int hf) {
  FragH f;
#pragma unroll
  for (int g = 0; g < 8; ++g) {
    int k = kofs + ((g & 3) << 1) + ((g & 4) << 2) + (hf << 3);
    f.u[g] = *(const unsigned*)(colp + k);
  }
  return f.h;
}

// Convert float4 -> 4 f16, one 8-byte LDS store (dst 8B-aligned).
__device__ __forceinline__ void cvt_store4(_Float16* dst, float4 v) {
  union { _Float16 h[4]; uint2 u; } tmp;
  tmp.h[0] = (_Float16)v.x; tmp.h[1] = (_Float16)v.y;
  tmp.h[2] = (_Float16)v.z; tmp.h[3] = (_Float16)v.w;
  *(uint2*)dst = tmp.u;
}

// Pack two f32 (consecutive k of one column) -> one dword LDS store.
__device__ __forceinline__ void cvt_store_pair(_Float16* dst, float a, float b) {
  union { _Float16 h[2]; unsigned u; } tmp;
  tmp.h[0] = (_Float16)a; tmp.h[1] = (_Float16)b;
  *(unsigned*)dst = tmp.u;
}

// ---------------------------------------------------------------------------
// GEMM: out[M,N] = A[M,K] * W[K,N] + bias[N] (+resid) (gelu?)   all fp32 global
// Block tile 128x64, BK=32, 256 threads = 8 waves; double-buffered LDS.
// Requires M%128==0, N%64==0, K%32==0 (true for all shapes here).
// ---------------------------------------------------------------------------
__global__ __launch_bounds__(256) void gemm_bias_kernel(
    const float* __restrict__ A, const float* __restrict__ W,
    const float* __restrict__ bias, const float* __restrict__ resid,
    float* __restrict__ out, int M, int N, int K, int act_gelu)
{
  __shared__ __align__(16) _Float16 As[2][128][40];   // [m][k], pad 8
  __shared__ __align__(16) _Float16 BsT[2][64][40];   // [n][k], pad 8
  const int t = threadIdx.x;
  const int wave = t >> 5, lane = t & 31;
  const int hf = lane >> 4, r = lane & 15;
  const int row0 = blockIdx.y * 128, col0 = blockIdx.x * 64;

  // A staging: 4 x float4 per thread (row-major).
  int arow[4], acol[4]; const float* aptr[4];
#pragma unroll
  for (int i = 0; i < 4; ++i) {
    int idx4 = t + 256 * i;
    arow[i] = idx4 >> 3; acol[i] = (idx4 & 7) << 2;
    aptr[i] = A + (size_t)(row0 + arow[i]) * K + acol[i];
  }
  // W staging: each thread owns a 2(k) x 4(n) micro-tile of the 32x64 W tile.
  const int wk = (t >> 4) << 1;       // 0,2,...,30
  const int wn = (t & 15) << 2;       // 0,4,...,60
  const float* wptr0 = W + (size_t)wk * N + col0 + wn;
  const float* wptr1 = wptr0 + N;

  v8f acc[4];
#pragma unroll
  for (int c = 0; c < 4; ++c)
#pragma unroll
    for (int i = 0; i < 8; ++i) acc[c][i] = 0.0f;

  float4 areg[4], wreg0, wreg1;
  // Prologue: load + store k-tile 0 into buffer 0.
#pragma unroll
  for (int i = 0; i < 4; ++i) areg[i] = *(const float4*)(aptr[i]);
  wreg0 = *(const float4*)(wptr0);
  wreg1 = *(const float4*)(wptr1);
#pragma unroll
  for (int i = 0; i < 4; ++i) cvt_store4(&As[0][arow[i]][acol[i]], areg[i]);
#pragma unroll
  for (int j = 0; j < 4; ++j)
    cvt_store_pair(&BsT[0][wn + j][wk], (&wreg0.x)[j], (&wreg1.x)[j]);

  const int nk = K >> 5;
  for (int kt = 0; kt < nk; ++kt) {
    const int buf = kt & 1;
    if (kt + 1 < nk) {                           // prefetch next tile to regs
      const int k0 = (kt + 1) << 5;
#pragma unroll
      for (int i = 0; i < 4; ++i) areg[i] = *(const float4*)(aptr[i] + k0);
      wreg0 = *(const float4*)(wptr0 + (size_t)k0 * N);
      wreg1 = *(const float4*)(wptr1 + (size_t)k0 * N);
    }
    __syncthreads();                             // buf tile ready for all waves
    v16h afrag = load_frag_a(&As[buf][wave * 16 + r][0], 0, hf);
    v16h bfr[4];
#pragma unroll
    for (int c = 0; c < 4; ++c) bfr[c] = load_frag_b(&BsT[buf][c * 16 + r][0], 0, hf);
#pragma unroll
    for (int c = 0; c < 4; ++c)
      acc[c] = __builtin_amdgcn_wmma_f32_16x16x32_f16(
          false, afrag, false, bfr[c], (short)0, acc[c], false, false);
    if (kt + 1 < nk) {                           // stage next tile to other buf
      const int nbuf = buf ^ 1;
#pragma unroll
      for (int i = 0; i < 4; ++i) cvt_store4(&As[nbuf][arow[i]][acol[i]], areg[i]);
#pragma unroll
      for (int j = 0; j < 4; ++j)
        cvt_store_pair(&BsT[nbuf][wn + j][wk], (&wreg0.x)[j], (&wreg1.x)[j]);
    }
  }

#pragma unroll
  for (int c = 0; c < 4; ++c) {
    int col = col0 + c * 16 + r;
    float bv = bias[col];
#pragma unroll
    for (int i = 0; i < 8; ++i) {
      int row = row0 + wave * 16 + i + hf * 8;
      float v = acc[c][i] + bv;
      if (resid) v += resid[(size_t)row * N + col];
      if (act_gelu) {
        float x = v;
        v = 0.5f * x * (1.0f + tanhf(0.7978845608028654f * (x + 0.044715f * x * x * x)));
      }
      out[(size_t)row * N + col] = v;
    }
  }
}

// ---------------------------------------------------------------------------
// LayerNorm: one block per row of 1024; y = (x-m)*rsqrt(var+eps)*g + b
// ---------------------------------------------------------------------------
__global__ __launch_bounds__(256) void ln_kernel(
    const float* __restrict__ x, const float* __restrict__ gamma,
    const float* __restrict__ beta, float* __restrict__ y)
{
  __shared__ float s1[256];
  __shared__ float s2[256];
  const int row = blockIdx.x, t = threadIdx.x;
  const float4 a = *(const float4*)(x + (size_t)row * HDIM + t * 4);
  s1[t] = a.x + a.y + a.z + a.w;
  s2[t] = a.x * a.x + a.y * a.y + a.z * a.z + a.w * a.w;
  __syncthreads();
  for (int o = 128; o > 0; o >>= 1) {
    if (t < o) { s1[t] += s1[t + o]; s2[t] += s2[t + o]; }
    __syncthreads();
  }
  const float m = s1[0] * (1.0f / HDIM);
  const float var = s2[0] * (1.0f / HDIM) - m * m;
  const float inv = rsqrtf(var + 1e-12f);
  const float4 g = *(const float4*)(gamma + t * 4);
  const float4 b = *(const float4*)(beta + t * 4);
  float4 o4;
  o4.x = (a.x - m) * inv * g.x + b.x;
  o4.y = (a.y - m) * inv * g.y + b.y;
  o4.z = (a.z - m) * inv * g.z + b.z;
  o4.w = (a.w - m) * inv * g.w + b.w;
  *(float4*)(y + (size_t)row * HDIM + t * 4) = o4;
}

// ---------------------------------------------------------------------------
// Attention: q,k,v,out are [B,S,H]; one block = 128 q rows of one (batch,head);
// flash-style online softmax over 32-wide KV tiles, double-buffered K/V staging.
// grid.x = (S/128) * NH * B = 256
// ---------------------------------------------------------------------------
__global__ __launch_bounds__(256) void attn_kernel(
    const float* __restrict__ q, const float* __restrict__ kg,
    const float* __restrict__ vg, const int* __restrict__ mask,
    float* __restrict__ out)
{
  __shared__ __align__(16) _Float16 Qs[128][72];      // [qrow][d]
  __shared__ __align__(16) _Float16 Ks[2][32][72];    // [kv][d]
  __shared__ __align__(16) _Float16 VsT[2][64][40];   // [d][kv]
  __shared__ __align__(16) _Float16 Ps[8][16][40];    // per-wave P tile [m][kv]
  __shared__ float ma[2][32];

  const int bid = blockIdx.x;
  const int qt = bid & 3;
  const int hd = (bid >> 2) & (NHEAD - 1);
  const int bb = bid >> 6;
  const int t = threadIdx.x;
  const int wave = t >> 5, lane = t & 31;
  const int hf = lane >> 4, r = lane & 15;
  const size_t base = ((size_t)bb * SLEN) * HDIM + hd * DHEAD;
  const int q0 = qt * 128;
  const float scale = 0.125f;  // 1/sqrt(64)

  // Stage Q tile (128x64) once: 8 x float4 per thread.
#pragma unroll
  for (int i = 0; i < 8; ++i) {
    int idx4 = t + 256 * i;
    int qr = idx4 >> 4, qc = (idx4 & 15) << 2;
    float4 vq = *(const float4*)(q + base + (size_t)(q0 + qr) * HDIM + qc);
    cvt_store4(&Qs[qr][qc], vq);
  }

  // K staging: 2 x float4 per thread (row-major).
  int krow[2], kcol[2]; const float* kptr[2];
#pragma unroll
  for (int i = 0; i < 2; ++i) {
    int idx4 = t + 256 * i;
    krow[i] = idx4 >> 4; kcol[i] = (idx4 & 15) << 2;
    kptr[i] = kg + base + (size_t)krow[i] * HDIM + kcol[i];
  }
  // V staging (transposed): each thread owns a 2(kv) x 4(d) micro-tile.
  const int vk = (t >> 4) << 1;       // kv pair base: 0,2,...,30
  const int vn = (t & 15) << 2;       // d quad: 0,4,...,60
  const float* vptr0 = vg + base + (size_t)vk * HDIM + vn;
  const float* vptr1 = vptr0 + HDIM;

  float4 kreg[2], vreg0, vreg1;
  int mreg = 1;
  // Prologue: stage KV tile 0 into buffer 0.
#pragma unroll
  for (int i = 0; i < 2; ++i) kreg[i] = *(const float4*)(kptr[i]);
  vreg0 = *(const float4*)(vptr0);
  vreg1 = *(const float4*)(vptr1);
  if (t < 32) mreg = mask[bb * SLEN + t];
#pragma unroll
  for (int i = 0; i < 2; ++i) cvt_store4(&Ks[0][krow[i]][kcol[i]], kreg[i]);
#pragma unroll
  for (int j = 0; j < 4; ++j)
    cvt_store_pair(&VsT[0][vn + j][vk], (&vreg0.x)[j], (&vreg1.x)[j]);
  if (t < 32) ma[0][t] = -10000.0f * (float)(1 - mreg);

  __syncthreads();   // Q (and KV tile 0) staged
  // Q fragments are loop-invariant: hoist.
  const v16h qa0 = load_frag_a(&Qs[wave * 16 + r][0], 0, hf);
  const v16h qa1 = load_frag_a(&Qs[wave * 16 + r][0], 32, hf);

  v8f acc[4];
  float mrun[8], lrun[8];
#pragma unroll
  for (int i = 0; i < 8; ++i) { mrun[i] = -1e30f; lrun[i] = 0.0f; }
#pragma unroll
  for (int c = 0; c < 4; ++c)
#pragma unroll
    for (int i = 0; i < 8; ++i) acc[c][i] = 0.0f;

  const int nkv = SLEN / 32;   // 16
  for (int kvt = 0; kvt < nkv; ++kvt) {
    const int buf = kvt & 1;
    if (kvt + 1 < nkv) {       // prefetch next KV tile to regs
      const size_t ofs = (size_t)(kvt + 1) * 32 * HDIM;
#pragma unroll
      for (int i = 0; i < 2; ++i) kreg[i] = *(const float4*)(kptr[i] + ofs);
      vreg0 = *(const float4*)(vptr0 + ofs);
      vreg1 = *(const float4*)(vptr1 + ofs);
      if (t < 32) mreg = mask[bb * SLEN + (kvt + 1) * 32 + t];
    }
    if (kvt > 0) __syncthreads();   // buf tile stores (prev iter) visible

    // scores S = Q * K^T  (contraction over d=64 in two 32-chunks)
    v8f sacc[2];
#pragma unroll
    for (int c = 0; c < 2; ++c)
#pragma unroll
      for (int i = 0; i < 8; ++i) sacc[c][i] = 0.0f;
    v16h kb[4];
#pragma unroll
    for (int c = 0; c < 2; ++c) {
      kb[2 * c]     = load_frag_b(&Ks[buf][c * 16 + r][0], 0, hf);
      kb[2 * c + 1] = load_frag_b(&Ks[buf][c * 16 + r][0], 32, hf);
    }
#pragma unroll
    for (int c = 0; c < 2; ++c) {
      sacc[c] = __builtin_amdgcn_wmma_f32_16x16x32_f16(
          false, qa0, false, kb[2 * c], (short)0, sacc[c], false, false);
      sacc[c] = __builtin_amdgcn_wmma_f32_16x16x32_f16(
          false, qa1, false, kb[2 * c + 1], (short)0, sacc[c], false, false);
    }

    // online softmax; row m = i + 8*hf lives across the 16 lanes of a half
#pragma unroll
    for (int i = 0; i < 8; ++i) {
      float s0 = sacc[0][i] * scale + ma[buf][r];
      float s1 = sacc[1][i] * scale + ma[buf][16 + r];
      float mx = fmaxf(s0, s1);
#pragma unroll
      for (int off = 1; off < 16; off <<= 1) mx = fmaxf(mx, __shfl_xor(mx, off, 32));
      float mnew = fmaxf(mrun[i], mx);
      float corr = __expf(mrun[i] - mnew);
      float p0 = __expf(s0 - mnew);
      float p1 = __expf(s1 - mnew);
      float ps = p0 + p1;
#pragma unroll
      for (int off = 1; off < 16; off <<= 1) ps += __shfl_xor(ps, off, 32);
      lrun[i] = lrun[i] * corr + ps;
      mrun[i] = mnew;
#pragma unroll
      for (int c = 0; c < 4; ++c) acc[c][i] *= corr;
      int mrow = i + hf * 8;
      Ps[wave][mrow][r]      = (_Float16)p0;
      Ps[wave][mrow][16 + r] = (_Float16)p1;
    }
    asm volatile("s_wait_dscnt 0" ::: "memory");  // wave-local LDS RAW fence

    // O += P * V  (contraction over 32 kv)
    v16h pa = load_frag_a(&Ps[wave][r][0], 0, hf);
    v16h vb[4];
#pragma unroll
    for (int d4 = 0; d4 < 4; ++d4) vb[d4] = load_frag_b(&VsT[buf][d4 * 16 + r][0], 0, hf);
#pragma unroll
    for (int d4 = 0; d4 < 4; ++d4)
      acc[d4] = __builtin_amdgcn_wmma_f32_16x16x32_f16(
          false, pa, false, vb[d4], (short)0, acc[d4], false, false);

    if (kvt + 1 < nkv) {      // stage next KV tile to other buffer
      const int nbuf = buf ^ 1;
#pragma unroll
      for (int i = 0; i < 2; ++i) cvt_store4(&Ks[nbuf][krow[i]][kcol[i]], kreg[i]);
#pragma unroll
      for (int j = 0; j < 4; ++j)
        cvt_store_pair(&VsT[nbuf][vn + j][vk], (&vreg0.x)[j], (&vreg1.x)[j]);
      if (t < 32) ma[nbuf][t] = -10000.0f * (float)(1 - mreg);
    }
  }

#pragma unroll
  for (int d4 = 0; d4 < 4; ++d4)
#pragma unroll
    for (int i = 0; i < 8; ++i) {
      int row = q0 + wave * 16 + i + hf * 8;
      out[base + (size_t)row * HDIM + d4 * 16 + r] = acc[d4][i] / lrun[i];
    }
}

// ---------------------------------------------------------------------------
// Host orchestration
// ---------------------------------------------------------------------------
static inline void launch_gemm(const float* A, const float* W, const float* bias,
                               const float* resid, float* out, int M, int N, int K,
                               int gelu, hipStream_t stream) {
  dim3 grid(N / 64, M / 128);
  gemm_bias_kernel<<<grid, 256, 0, stream>>>(A, W, bias, resid, out, M, N, K, gelu);
}

extern "C" void kernel_launch(void* const* d_in, const int* in_sizes, int n_in,
                              void* d_out, int out_size, void* d_ws, size_t ws_size,
                              hipStream_t stream) {
  (void)in_sizes; (void)n_in; (void)out_size; (void)ws_size;
  const int H = HDIM, F = FDIM, Mt = MTOK;

  const float* hidden   = (const float*)d_in[0];
  const float* enc      = (const float*)d_in[1];
  const int*   amask    = (const int*)d_in[2];
  const int*   emask    = (const int*)d_in[3];
  const float* sa_qkv_w = (const float*)d_in[4];
  const float* sa_qkv_b = (const float*)d_in[5];
  const float* sa_out_w = (const float*)d_in[6];
  const float* sa_out_b = (const float*)d_in[7];
  const float* sa_ln    = (const float*)d_in[8];
  const float* ca_qkv_w = (const float*)d_in[9];
  const float* ca_qkv_b = (const float*)d_in[10];
  const float* ca_out_w = (const float*)d_in[11];
  const float* ca_out_b = (const float*)d_in[12];
  const float* ca_ln    = (const float*)d_in[13];
  const float* o1_w     = (const float*)d_in[14];
  const float* o1_b     = (const float*)d_in[15];
  const float* o1_ln    = (const float*)d_in[16];
  const float* ffn_w    = (const float*)d_in[17];
  const float* ffn_b    = (const float*)d_in[18];
  const float* o2_w     = (const float*)d_in[19];
  const float* o2_b     = (const float*)d_in[20];
  const float* o2_ln    = (const float*)d_in[21];

  // workspace: 8 fp32 [Mt,H] buffers + one [Mt,F] = ~96 MB
  float* p = (float*)d_ws;
  float* h_buf    = p; p += (size_t)Mt * H;
  float* qq       = p; p += (size_t)Mt * H;
  float* kk       = p; p += (size_t)Mt * H;
  float* vv       = p; p += (size_t)Mt * H;
  float* vec      = p; p += (size_t)Mt * H;
  float* tmp      = p; p += (size_t)Mt * H;
  float* attn_out = p; p += (size_t)Mt * H;
  float* cross_out= p; p += (size_t)Mt * H;
  float* inter    = p; p += (size_t)Mt * F;

  hipMemcpyAsync(h_buf, hidden, (size_t)Mt * H * sizeof(float),
                 hipMemcpyDeviceToDevice, stream);

  const int ATT_GRID = (SLEN / 128) * NHEAD * BSZ;   // 256

  for (int i = 0; i < LBL; ++i) {
    // ---- self attention ----
    const float* w  = sa_qkv_w + (size_t)i * 3 * H * H;
    const float* bq = sa_qkv_b + (size_t)i * 3 * H;
    launch_gemm(h_buf, w,                 bq,       nullptr, qq, Mt, H, H, 0, stream);
    launch_gemm(h_buf, w + (size_t)H*H,   bq + H,   nullptr, kk, Mt, H, H, 0, stream);
    launch_gemm(h_buf, w + (size_t)2*H*H, bq + 2*H, nullptr, vv, Mt, H, H, 0, stream);
    attn_kernel<<<ATT_GRID, 256, 0, stream>>>(qq, kk, vv, amask, vec);
    launch_gemm(vec, sa_out_w + (size_t)i*H*H, sa_out_b + (size_t)i*H, h_buf, tmp,
                Mt, H, H, 0, stream);
    ln_kernel<<<Mt, 256, 0, stream>>>(tmp, sa_ln + (size_t)i*2*H,
                                      sa_ln + (size_t)i*2*H + H, attn_out);

    // ---- cross attention (residual = h, per reference) ----
    const float* cw = ca_qkv_w + (size_t)i * 3 * H * H;
    const float* cb = ca_qkv_b + (size_t)i * 3 * H;
    launch_gemm(h_buf, cw,                 cb,       nullptr, qq, Mt, H, H, 0, stream);
    launch_gemm(enc,   cw + (size_t)H*H,   cb + H,   nullptr, kk, Mt, H, H, 0, stream);
    launch_gemm(enc,   cw + (size_t)2*H*H, cb + 2*H, nullptr, vv, Mt, H, H, 0, stream);
    attn_kernel<<<ATT_GRID, 256, 0, stream>>>(qq, kk, vv, emask, vec);
    launch_gemm(vec, ca_out_w + (size_t)i*H*H, ca_out_b + (size_t)i*H, h_buf, tmp,
                Mt, H, H, 0, stream);
    ln_kernel<<<Mt, 256, 0, stream>>>(tmp, ca_ln + (size_t)i*2*H,
                                      ca_ln + (size_t)i*2*H + H, cross_out);

    // ---- output_1: dense(attn_out) + cross_out, LN ----
    launch_gemm(attn_out, o1_w + (size_t)i*H*H, o1_b + (size_t)i*H, cross_out, tmp,
                Mt, H, H, 0, stream);
    ln_kernel<<<Mt, 256, 0, stream>>>(tmp, o1_ln + (size_t)i*2*H,
                                      o1_ln + (size_t)i*2*H + H, attn_out);

    // ---- FFN (gelu_new) + output_2 ----
    launch_gemm(attn_out, ffn_w + (size_t)i*H*F, ffn_b + (size_t)i*F, nullptr, inter,
                Mt, F, H, 1, stream);
    launch_gemm(inter, o2_w + (size_t)i*F*H, o2_b + (size_t)i*H, attn_out, tmp,
                Mt, H, F, 0, stream);
    ln_kernel<<<Mt, 256, 0, stream>>>(tmp, o2_ln + (size_t)i*2*H,
                                      o2_ln + (size_t)i*2*H + H, h_buf);
  }

  hipMemcpyAsync(d_out, h_buf, (size_t)Mt * H * sizeof(float),
                 hipMemcpyDeviceToDevice, stream);
}
